// OTPE_pass_67181878444192
// MI455X (gfx1250) — matrix operand order repaired and checked by reading, not compiled, not run on hardware
//
#include <hip/hip_runtime.h>
#include <math.h>

typedef __attribute__((ext_vector_type(16))) _Float16 v16h;
typedef __attribute__((ext_vector_type(8)))  _Float16 v8h;
typedef __attribute__((ext_vector_type(4)))  _Float16 v4h;
typedef __attribute__((ext_vector_type(8)))  float    v8f;
typedef __attribute__((ext_vector_type(4)))  float    v4f;

#define NDIM 4096
// sigmoid(2.0)
#define SIG_TAU 0.88079707797788231f
#define VTH 1.0f

// ---------------------------------------------------------------------------
// Kernel 1: I = x@W + b via WMMA (f16 hi/lo split => ~f32 accuracy), plus all
// vector-sized outputs. One wave32 per 16-column strip (256 strips total).
//
// x is staged once per block into LDS, pre-split into f16 hi/lo, with a
// 32-half zero pad at [NDIM..NDIM+31]. Lanes that supply the zero rows of A
// (everything except lanes 0 and 16) read the pad: a single index cndmask
// replaces the predicated-load mess of the previous version.
//
// A fragment layout (16x32 f16): lanes 0-15 hold K 0-15 (e=0..15);
// lanes 16-31 hold K 16-31. Row 0 only.
// B fragment layout (32x16 f16): lane n<16 -> col n, K 0-15; lane n>=16 ->
// col n-16, K 16-31. D row 0 = lanes 0-15, c[0].
// ---------------------------------------------------------------------------
__global__ void __launch_bounds__(256)
otpe_matvec_wmma(const float* __restrict__ x,
                 const float* __restrict__ u,
                 const float* __restrict__ W,
                 const float* __restrict__ b,
                 const float* __restrict__ J_u_b,
                 const float* __restrict__ r2_b,
                 const float* __restrict__ J_u_x,
                 const float* __restrict__ ratio,
                 float* __restrict__ out_s,
                 float* __restrict__ out_u,
                 float* __restrict__ out_Jub,
                 float* __restrict__ out_r2b,
                 float* __restrict__ out_Jux,
                 float* __restrict__ out_ratio,
                 float* __restrict__ sg_ws)
{
    __shared__ __align__(16) _Float16 xh[NDIM + 32];
    __shared__ __align__(16) _Float16 xl[NDIM + 32];

    // ---- cooperative pre-split of x into LDS (4 iterations of v4) ---------
    for (int t = threadIdx.x * 4; t < NDIM; t += blockDim.x * 4) {
        const v4f xv = *(const v4f*)(x + t);
        v4h hh, ll;
        #pragma unroll
        for (int e = 0; e < 4; ++e) {
            const _Float16 h = (_Float16)xv[e];
            hh[e] = h;
            ll[e] = (_Float16)(xv[e] - (float)h);
        }
        *(v4h*)(xh + t) = hh;
        *(v4h*)(xl + t) = ll;
    }
    // zero pad (read by the lanes that feed zero rows of A)
    if (threadIdx.x < 8) {
        v4h z;
        #pragma unroll
        for (int e = 0; e < 4; ++e) z[e] = (_Float16)0.0f;
        *(v4h*)(xh + NDIM + threadIdx.x * 4) = z;
        *(v4h*)(xl + NDIM + threadIdx.x * 4) = z;
    }
    __syncthreads();

    const int lane    = threadIdx.x & 31;
    const int wave    = threadIdx.x >> 5;
    const int strip   = blockIdx.x * (blockDim.x >> 5) + wave;  // 0..255
    const int j0      = strip << 4;
    const int laneN   = lane & 15;
    const int halfsel = lane >> 4;
    const bool rowlane = laneN == 0;          // lanes 0 and 16 carry A row 0

    v8f c;
    #pragma unroll
    for (int e = 0; e < 8; ++e) c[e] = 0.0f;

    const float* wcol = W + (size_t)(halfsel << 4) * NDIM + (j0 + laneN);

    for (int k0 = 0; k0 < NDIM; k0 += 32) {
        // ---- A fragment: 4 unconditional ds_load_b128 + index select ------
        const int kb = rowlane ? (k0 + (halfsel << 4)) : NDIM;   // pad if zero
        const v8h h0 = *(const v8h*)(xh + kb);
        const v8h h1 = *(const v8h*)(xh + kb + 8);
        const v8h l0 = *(const v8h*)(xl + kb);
        const v8h l1 = *(const v8h*)(xl + kb + 8);
        const v16h a_hi = __builtin_shufflevector(h0, h1,
            0,1,2,3,4,5,6,7,8,9,10,11,12,13,14,15);
        const v16h a_lo = __builtin_shufflevector(l0, l1,
            0,1,2,3,4,5,6,7,8,9,10,11,12,13,14,15);

        // ---- B fragment: W[k0+K][j0+n], hi/lo split -----------------------
        const float* wp = wcol + (size_t)k0 * NDIM;
        v16h b_hi, b_lo;
        #pragma unroll
        for (int e = 0; e < 16; ++e) {
            const float wv = wp[(size_t)e * NDIM];
            const _Float16 h = (_Float16)wv;
            b_hi[e] = h;
            b_lo[e] = (_Float16)(wv - (float)h);
        }

        // ---- 3-term split product: hi*hi + hi*lo + lo*hi ------------------
        c = __builtin_amdgcn_wmma_f32_16x16x32_f16(false, a_hi, false, b_hi,
                                                   (short)0, c, false, false);
        c = __builtin_amdgcn_wmma_f32_16x16x32_f16(false, a_hi, false, b_lo,
                                                   (short)0, c, false, false);
        c = __builtin_amdgcn_wmma_f32_16x16x32_f16(false, a_lo, false, b_hi,
                                                   (short)0, c, false, false);
    }

    // scalar recursion on ratio (pure function of input -> deterministic)
    const float r_old     = SIG_TAU * ratio[0];
    const float ratio_new = r_old + 1.0f;
    const float r         = r_old / ratio_new;

    if (lane < 16) {
        const int   j     = j0 + lane;
        const float I     = c[0] + b[j];
        const float u_new = SIG_TAU * u[j] + I;
        const float d     = fabsf(u_new - VTH) + 1.0f;
        const float sg    = 1.0f / (d * d);
        const float s     = (u_new >= VTH) ? 1.0f : 0.0f;

        out_s[j] = s;
        out_u[j] = u_new - s * VTH;

        // J_s_b = grad_s_u*J_u_b + sg = sg*(sig_tau*J_u_b + 1) = sg*J_u_b_new
        const float jub_new = SIG_TAU * J_u_b[j] + 1.0f;
        out_Jub[j] = jub_new;
        out_r2b[j] = r * r2_b[j] + (1.0f - r) * (sg * jub_new);
        out_Jux[j] = r * J_u_x[j] + (1.0f - r) * sg;   // grad_s_x == sg
        sg_ws[j]   = sg;
    }
    if (blockIdx.x == 0 && threadIdx.x == 0) out_ratio[0] = ratio_new;
}

// ---------------------------------------------------------------------------
// Kernel 2: fused elementwise stream over the four matrix recursions.
// Per float4: 5 loads (80B) + 4 stores (64B); bandwidth-bound by design
// (~576 MB total => ~25 us at 23.3 TB/s).
// Fusions:  J_s_W  = sg * J_u_W_new        (since grad_s_u = sg*sig_tau)
//           s_pass = sg * u_pass_new       (since u_pass_new = sig_tau*u_pass+W)
// ---------------------------------------------------------------------------
__global__ void __launch_bounds__(256)
otpe_stream(const float* __restrict__ x,
            const float* __restrict__ W,
            const float* __restrict__ J_u_W,
            const float* __restrict__ r2_W,
            const float* __restrict__ pass_mat,
            const float* __restrict__ u_pass,
            const float* __restrict__ ratio,
            const float* __restrict__ sg_ws,
            float* __restrict__ out_JuW,
            float* __restrict__ out_r2W,
            float* __restrict__ out_pass,
            float* __restrict__ out_upass)
{
    const size_t idx = (size_t)blockIdx.x * blockDim.x + threadIdx.x; // /float4
    const size_t off = idx << 2;
    const int    i   = (int)(idx >> 10);            // row   (1024 float4 / row)
    const int    j0  = (int)((idx & 1023) << 2);    // column of first element

    const float r_old   = SIG_TAU * ratio[0];
    const float r       = r_old / (r_old + 1.0f);
    const float one_m_r = 1.0f - r;
    const float xi      = x[i];
    const v4f   sg      = *(const v4f*)(sg_ws + j0);

    const v4f w  = *(const v4f*)(W + off);          // RT: may still sit in 192MB L2
    const v4f ju = __builtin_nontemporal_load((const v4f*)(J_u_W    + off));
    const v4f r2 = __builtin_nontemporal_load((const v4f*)(r2_W     + off));
    const v4f pm = __builtin_nontemporal_load((const v4f*)(pass_mat + off));
    const v4f up = __builtin_nontemporal_load((const v4f*)(u_pass   + off));

    v4f jn, r2n, un, pn;
    #pragma unroll
    for (int e = 0; e < 4; ++e) {
        jn[e]  = __builtin_fmaf(SIG_TAU, ju[e], xi);                 // J_u_W_new
        r2n[e] = __builtin_fmaf(r, r2[e], one_m_r * sg[e] * jn[e]);  // r2_W_new
        un[e]  = __builtin_fmaf(SIG_TAU, up[e], w[e]);               // u_pass_new
        pn[e]  = __builtin_fmaf(SIG_TAU, pm[e], sg[e] * un[e]);      // pass_new
    }
    __builtin_nontemporal_store(jn,  (v4f*)(out_JuW   + off));
    __builtin_nontemporal_store(r2n, (v4f*)(out_r2W   + off));
    __builtin_nontemporal_store(pn,  (v4f*)(out_pass  + off));
    __builtin_nontemporal_store(un,  (v4f*)(out_upass + off));
}

extern "C" void kernel_launch(void* const* d_in, const int* in_sizes, int n_in,
                              void* d_out, int out_size, void* d_ws, size_t ws_size,
                              hipStream_t stream) {
    const float* x        = (const float*)d_in[0];
    const float* u        = (const float*)d_in[1];
    const float* W        = (const float*)d_in[2];
    const float* b        = (const float*)d_in[3];
    const float* J_u_W    = (const float*)d_in[4];
    const float* J_u_b    = (const float*)d_in[5];
    const float* r2_W     = (const float*)d_in[6];
    const float* r2_b     = (const float*)d_in[7];
    const float* J_u_x    = (const float*)d_in[8];
    const float* pass_mat = (const float*)d_in[9];
    const float* u_pass   = (const float*)d_in[10];
    const float* ratio    = (const float*)d_in[11];

    float* out = (float*)d_out;
    const size_t NN = (size_t)NDIM * NDIM;
    float* out_s     = out;                    // [4096]
    float* out_u     = out_s     + NDIM;       // [4096]
    float* out_JuW   = out_u     + NDIM;       // [4096,4096]
    float* out_Jub   = out_JuW   + NN;         // [4096]
    float* out_r2W   = out_Jub   + NDIM;       // [4096,4096]
    float* out_r2b   = out_r2W   + NN;         // [4096]
    float* out_Jux   = out_r2b   + NDIM;       // [4096]
    float* out_pass  = out_Jux   + NDIM;       // [4096,4096]
    float* out_upass = out_pass  + NN;         // [4096,4096]
    float* out_ratio = out_upass + NN;         // [1]

    float* sg_ws = (float*)d_ws;               // 4096 floats

    // 256 strips of 16 columns; 8 waves/block -> 32 blocks
    otpe_matvec_wmma<<<32, 256, 0, stream>>>(
        x, u, W, b, J_u_b, r2_b, J_u_x, ratio,
        out_s, out_u, out_Jub, out_r2b, out_Jux, out_ratio, sg_ws);

    // 16.8M elements / 4 per thread / 256 per block = 16384 blocks
    otpe_stream<<<(int)(NN / 4 / 256), 256, 0, stream>>>(
        x, W, J_u_W, r2_W, pass_mat, u_pass, ratio, sg_ws,
        out_JuW, out_r2W, out_pass, out_upass);
}
